// RNO_2D_44770739093509
// MI455X (gfx1250) — matrix operand order repaired and checked by reading, compiled, not run on hardware
//
#include <hip/hip_runtime.h>
#include <math.h>
#include <stdint.h>

// ---------------------------------------------------------------------------
// RNO-2D forward for MI455X (gfx1250, wave32, WMMA + TDM).
// All dense math is bf16 WMMA GEMMs with f32 accumulation; A-operands are
// prepacked once per launch into the WMMA A lane-swizzle (2x b128 per tile).
// B-operand tiles are staged into LDS by the Tensor Data Mover
// (tensor_load_to_lds, TENSORcnt) with TDM-applied row padding so the
// ds_load side is bank-conflict free.
// ---------------------------------------------------------------------------

typedef __attribute__((ext_vector_type(16))) __bf16 v16bf;
typedef __attribute__((ext_vector_type(8)))  float  v8f;
typedef __attribute__((ext_vector_type(4)))  unsigned int u32x4;
typedef __attribute__((ext_vector_type(8)))  int i32x8;
typedef __attribute__((ext_vector_type(4)))  int i32x4;

#define WMMA_BF16(a, b, c) \
  __builtin_amdgcn_wmma_f32_16x16x32_bf16(false, (a), false, (b), (short)0, (c), false, false)

#if defined(__HIP_DEVICE_COMPILE__) && defined(__has_builtin)
#  if __has_builtin(__builtin_amdgcn_tensor_load_to_lds)
#    define HAVE_TDM 1
#  endif
#endif
#ifndef HAVE_TDM
#  define HAVE_TDM 0
#endif

// Problem constants
#define NB 8
#define NT 16
#define NC 64
#define SS 64
#define SP (SS*SS)            // 4096 spatial points
#define TS (NB*NC*SP)         // 2097152 floats per time-slot (b,c,p)
#define VTOT (NT*TS)          // 33554432 floats for the full sequence
#define S3PACK (512*3*4096)   // bf16 elems per fourier-param pack

// A-operand K mapping inside a 16x32 bf16 tile (contiguous: K = (lane>>4)*16 + e, matches B)
__device__ __forceinline__ int kmapA(int lane, int e) {
  return (lane >> 4) * 16 + e;
}

// ---------------------------------------------------------------------------
// TDM: issue a 2D tile load Global->LDS (one wave issues; EXEC ignored).
// Tile: tile_h rows x tile_w dwords, global row stride = stride dwords.
// pad_log encodes D# pad_interval so that 1 dword is inserted per row
// (row must be exactly (1<<pad_log)*2 dwords long) -> LDS row stride tile_w+1.
// D# built per ISA 8.3-8.5; groups 2/3 zero (2D tensor).
// ---------------------------------------------------------------------------
__device__ __forceinline__ void tdm_load_2d(float* lds_tile, const float* gsrc,
                                            unsigned tile_w, unsigned tile_h,
                                            unsigned stride, unsigned pad_log) {
#if HAVE_TDM
  // Low 32 bits of a flat LDS address are the LDS byte offset (aperture map).
  unsigned lds_off = (unsigned)(uintptr_t)lds_tile;
  unsigned long long ga = (unsigned long long)(uintptr_t)gsrc;
  u32x4 g0;
  g0[0] = 1u;                                        // count=1 valid, user mode
  g0[1] = lds_off;                                   // lds_addr (bytes)
  g0[2] = (unsigned)ga;                              // global_addr[31:0]
  g0[3] = (unsigned)((ga >> 32) & 0x01ffffffu) | (2u << 30);  // addr[56:32] | type=2
  i32x8 g1;
  g1[0] = (int)((2u << 16) | (1u << 20) | (pad_log << 22));   // 4B elems, pad_enable,
                                                              // pad_interval, pad_amount=0 (1 dw)
  g1[1] = (int)((tile_w & 0xffffu) << 16);                    // tensor_dim0 lo16 (=tile_w)
  g1[2] = (int)(((tile_w >> 16) & 0xffffu) | ((tile_h & 0xffffu) << 16)); // dim0 hi | dim1 lo
  g1[3] = (int)(((tile_h >> 16) & 0xffffu) | ((tile_w & 0xffffu) << 16)); // dim1 hi | tile_dim0
  g1[4] = (int)(tile_h & 0xffffu);                            // tile_dim1 | tile_dim2=0
  g1[5] = (int)stride;                                        // tensor_dim0_stride[31:0]
  g1[6] = 0;                                                  // stride hi | dim1_stride lo
  g1[7] = 0;
  i32x4 z = {0, 0, 0, 0};
#if __clang_major__ >= 23
  i32x8 z8 = {0, 0, 0, 0, 0, 0, 0, 0};
  __builtin_amdgcn_tensor_load_to_lds(g0, g1, z, z, z8, 0);
#else
  __builtin_amdgcn_tensor_load_to_lds(g0, g1, z, z, 0);
#endif
#else
  (void)lds_tile; (void)gsrc; (void)tile_w; (void)tile_h; (void)stride; (void)pad_log;
#endif
}

// ---------------------------------------------------------------------------
// Prep kernels: pack A operands into WMMA lane layout (bf16)
// ---------------------------------------------------------------------------

// DFT coefficient packs. stage: 1=DFT-y(32x64) 2=DFT-x(64x128)
// 4=iDFT-x(128x64) 5=iDFT-y(64x32). Layout [mt][ks][lane][e].
__global__ __launch_bounds__(256) void k_prep_dft(__bf16* __restrict__ out,
                                                  int stage, int nKs, int total) {
  int idx = blockIdx.x * 256 + threadIdx.x;
  if (idx >= total) return;
  int e = idx & 15, lane = (idx >> 4) & 31;
  int t = idx >> 9;
  int ks = t % nKs, mt = t / nKs;
  int row = mt * 16 + (lane & 15);
  int k = ks * 32 + kmapA(lane, e);
  const float W0 = 0.09817477042468103870195760572749f; // 2*pi/64
  float val = 0.f;
  if (stage == 1) {            // rows<16: cos (Tr); rows>=16: -sin (Ti); fwd norm 1/4096
    int n = row & 15;
    float th = W0 * (float)((n * k) & 63);
    val = ((row < 16) ? cosf(th) : -sinf(th)) * (1.0f / 4096.0f);
  } else if (stage == 2) {     // Z = sum_x e^{-i th} T ; K: [Tr x64 | Ti x64]
    int mi = row & 31; int m = (mi < 16) ? mi : (32 + mi);
    int xx = k & 63;
    float th = W0 * (float)((m * xx) & 63);
    bool tr = (k < 64);
    val = (row < 32) ? (tr ? cosf(th) : sinf(th))
                     : (tr ? -sinf(th) : cosf(th));
  } else if (stage == 4) {     // U = sum_mi e^{+i ph} O ; K: [Or x32 | Oi x32]
    int xx = row & 63; int mi = k & 31; int m = (mi < 16) ? mi : (32 + mi);
    float th = W0 * (float)((m * xx) & 63);
    bool orr = (k < 32);
    val = (row < 64) ? (orr ? cosf(th) : -sinf(th))
                     : (orr ? sinf(th) : cosf(th));
  } else {                     // y = sum_n cn*(cos*Ur - sin*Ui) ; K: [Ur x16 | Ui x16]
    int n = k & 15;
    float cn = (n == 0) ? 1.f : 2.f;
    float th = W0 * (float)((n * row) & 63);
    val = (k < 16) ? (cn * cosf(th)) : (-cn * sinf(th));
  }
  out[idx] = (__bf16)val;
}

// Pointwise conv weights: A[o,c] = cw[o*64+c].  Layout [mt4][ks2][lane][e].
__global__ __launch_bounds__(256) void k_prep_pw(const float* __restrict__ cw,
                                                 __bf16* __restrict__ out) {
  int idx = blockIdx.x * 256 + threadIdx.x;   // 4096 total
  int e = idx & 15, lane = (idx >> 4) & 31, ks = (idx >> 9) & 1, mt = idx >> 10;
  int o = mt * 16 + (lane & 15);
  int c = ks * 32 + kmapA(lane, e);
  out[idx] = (__bf16)cw[o * 64 + c];
}

// Mode-mix weights. Pack index (((md*3+which)*8 + mt*2 + ks)*32 + lane)*16 + e.
// which: 0 = Wr^T, 1 = Wi^T, 2 = -Wi^T.  w layout (i, o, m, n).
__global__ __launch_bounds__(256) void k_prep_s3(const float* __restrict__ swr1,
                                                 const float* __restrict__ swi1,
                                                 const float* __restrict__ swr2,
                                                 const float* __restrict__ swi2,
                                                 __bf16* __restrict__ out) {
  int idx = blockIdx.x * 256 + threadIdx.x;   // 6291456 total
  int e = idx & 15, lane = (idx >> 4) & 31;
  int t = idx >> 9;
  int ks = t & 1, mt = (t >> 1) & 3;
  int rest = t >> 3;
  int which = rest % 3, md = rest / 3;
  int mi = md >> 4, n = md & 15;
  int o = mt * 16 + (lane & 15);
  int i = ks * 32 + kmapA(lane, e);
  const float* wr = (mi < 16) ? swr1 : swr2;
  const float* wi = (mi < 16) ? swi1 : swi2;
  int m = mi & 15;
  int widx = ((i * 64 + o) * 16 + m) * 16 + n;
  float v = (which == 0) ? wr[widx] : ((which == 1) ? wi[widx] : -wi[widx]);
  out[idx] = (__bf16)v;
}

// ---------------------------------------------------------------------------
// WMMA GEMM kernels (block = 256 threads = 8 wave32 waves)
// B tile mapping (32x16 KxN): N col = lane&15, K = (lane>>4)*16 + e.
// C/D: col = lane&15, row = (lane>>4)*8 + j.
// ---------------------------------------------------------------------------

// Pointwise conv: Y[b,o,p] = sum_c cw[o,c]*X[b,c,p] + cb[o].
// TDM stages a 64(c) x 32(p) tile, global row stride 4096, LDS stride 33.
__global__ __launch_bounds__(256) void k_pw(const float* __restrict__ X,
                                            float* __restrict__ Y,
                                            const __bf16* __restrict__ Apk,
                                            const float* __restrict__ cb) {
  __shared__ float tile[64 * 33];
  int wave = threadIdx.x >> 5, lane = threadIdx.x & 31;
  int col0 = blockIdx.x * 32;                 // global column = b*4096 + p
  int b = col0 >> 12;
  int p0 = col0 & 4095;
  const float* src = X + (size_t)b * NC * SP + p0;
#if HAVE_TDM
  if (threadIdx.x < 32) {
    tdm_load_2d(tile, src, 32, 64, 4096, 4);
    __builtin_amdgcn_s_wait_tensorcnt(0);
  }
#else
#pragma unroll
  for (int i = 0; i < 8; ++i) {
    int idx = threadIdx.x + i * 256;
    tile[(idx >> 5) * 33 + (idx & 31)] = src[(idx >> 5) * SP + (idx & 31)];
  }
#endif
  __syncthreads();
  int mt = wave & 3, nt = wave >> 2;
  int pl = nt * 16 + (lane & 15);
  v8f acc = {};
#pragma unroll
  for (int ks = 0; ks < 2; ++ks) {
    v16bf a = *(const v16bf*)(Apk + ((mt * 2 + ks) * 32 + lane) * 16);
    int kb = ks * 32 + ((lane >> 4) << 4);
    v16bf bv;
#pragma unroll
    for (int e = 0; e < 16; ++e) bv[e] = (__bf16)tile[(kb + e) * 33 + pl];
    acc = WMMA_BF16(a, bv, acc);
  }
  float* Yb = Y + (size_t)b * NC * SP;
  int p = p0 + pl;
  int r0 = mt * 16 + ((lane >> 4) << 3);
#pragma unroll
  for (int j = 0; j < 8; ++j) {
    int o = r0 + j;
    Yb[o * SP + p] = acc[j] + cb[o];
  }
}

// S1: partial DFT along y. cols=(b*64+c)*64+x, T[(bc*16+n)*128 + ri*64 + x].
// TDM stages 64(cols) x 64(y) tile (contiguous region), LDS stride 65.
__global__ __launch_bounds__(256) void k_s1(const float* __restrict__ X,
                                            float* __restrict__ T,
                                            const __bf16* __restrict__ Apk) {
  __shared__ float tile[64 * 65];
  int wave = threadIdx.x >> 5, lane = threadIdx.x & 31;
  int col0 = blockIdx.x * 64;
  const float* src = X + (size_t)col0 * 64;
#if HAVE_TDM
  if (threadIdx.x < 32) {
    tdm_load_2d(tile, src, 64, 64, 64, 5);
    __builtin_amdgcn_s_wait_tensorcnt(0);
  }
#else
#pragma unroll
  for (int i = 0; i < 16; ++i) {
    int idx = threadIdx.x + i * 256;
    tile[(idx >> 6) * 65 + (idx & 63)] = src[idx];
  }
#endif
  __syncthreads();
  int mt = wave & 1, nt = wave >> 1;
  int cl = nt * 16 + (lane & 15);
  v8f acc = {};
#pragma unroll
  for (int ks = 0; ks < 2; ++ks) {
    v16bf a = *(const v16bf*)(Apk + ((mt * 2 + ks) * 32 + lane) * 16);
    int kb = ks * 32 + ((lane >> 4) << 4);
    v16bf bv;
#pragma unroll
    for (int e = 0; e < 16; ++e) bv[e] = (__bf16)tile[cl * 65 + kb + e];
    acc = WMMA_BF16(a, bv, acc);
  }
  int col = col0 + cl;
  int bc = col >> 6, xx = col & 63;
  int r0 = mt * 16 + ((lane >> 4) << 3);
#pragma unroll
  for (int j = 0; j < 8; ++j) {
    int mrow = r0 + j;
    int ri = mrow >> 4, n = mrow & 15;
    T[(bc * 16 + n) * 128 + ri * 64 + xx] = acc[j];
  }
}

// S2: partial DFT along x. One block per (b,n). K=128 over [Tr|Ti], N=c.
// TDM stages 64(c) x 128(k) tile, global row stride 2048, LDS stride 129.
__global__ __launch_bounds__(256) void k_s2(const float* __restrict__ T,
                                            float* __restrict__ Z,
                                            const __bf16* __restrict__ Apk) {
  __shared__ float tile[64 * 129];
  int wave = threadIdx.x >> 5, lane = threadIdx.x & 31;
  int b = blockIdx.x >> 4, n = blockIdx.x & 15;
  const float* Tb = T + b * 131072 + n * 128;   // + c*2048 + k
#if HAVE_TDM
  if (threadIdx.x < 32) {
    tdm_load_2d(tile, Tb, 128, 64, 2048, 6);
    __builtin_amdgcn_s_wait_tensorcnt(0);
  }
#else
#pragma unroll
  for (int i = 0; i < 32; ++i) {
    int idx = threadIdx.x + i * 256;
    tile[(idx >> 7) * 129 + (idx & 127)] = Tb[(idx >> 7) * 2048 + (idx & 127)];
  }
#endif
  __syncthreads();
#pragma unroll
  for (int r = 0; r < 2; ++r) {
    int wt = wave + r * 8;
    int mt = wt & 3, nt = wt >> 2;
    int c = nt * 16 + (lane & 15);
    v8f acc = {};
#pragma unroll
    for (int ks = 0; ks < 4; ++ks) {
      v16bf a = *(const v16bf*)(Apk + ((mt * 4 + ks) * 32 + lane) * 16);
      int kb = ks * 32 + ((lane >> 4) << 4);
      v16bf bv;
#pragma unroll
      for (int e = 0; e < 16; ++e) bv[e] = (__bf16)tile[c * 129 + kb + e];
      acc = WMMA_BF16(a, bv, acc);
    }
    int r0 = mt * 16 + ((lane >> 4) << 3);
#pragma unroll
    for (int j = 0; j < 8; ++j) {
      int row = r0 + j;
      int comp = row >> 5, mi = row & 31;
      Z[comp * 262144 + ((mi * 16 + n) * 8 + b) * 64 + c] = acc[j];
    }
  }
}

// S3: per-mode complex channel mix. One block per mode md=mi*16+n.
// out_r = Wr*Zr + (-Wi)*Zi ; out_i = Wi*Zr + Wr*Zi. N=b (8 valid of 16).
__global__ __launch_bounds__(256) void k_s3(const float* __restrict__ Z,
                                            float* __restrict__ O,
                                            const __bf16* __restrict__ Apk) {
  int wave = threadIdx.x >> 5, lane = threadIdx.x & 31;
  int md = blockIdx.x;
  int mi = md >> 4, n = md & 15;
  int comp = wave >> 2, mt = wave & 3;
  int colb = lane & 15;
  v8f acc = {};
#pragma unroll
  for (int term = 0; term < 2; ++term) {
    int which = (comp == 0) ? (term ? 2 : 0) : (term ? 0 : 1);
    const __bf16* Ap = Apk + (((size_t)(md * 3 + which) * 8) + mt * 2) * 512;
    const float* Zt = Z + term * 262144 + md * 512;   // + b*64 + c
#pragma unroll
    for (int ks = 0; ks < 2; ++ks) {
      v16bf a = *(const v16bf*)(Ap + (ks * 32 + lane) * 16);
      int kb = ks * 32 + ((lane >> 4) << 4);
      v16bf bv;
#pragma unroll
      for (int e = 0; e < 16; ++e)
        bv[e] = (colb < 8) ? (__bf16)Zt[colb * 64 + kb + e] : (__bf16)0.f;
      acc = WMMA_BF16(a, bv, acc);
    }
  }
  if (colb < 8) {
    int r0 = mt * 16 + ((lane >> 4) << 3);
#pragma unroll
    for (int j = 0; j < 8; ++j) {
      int o = r0 + j;
      O[((colb * 16 + n) * 64 + comp * 32 + mi) * 64 + o] = acc[j];
    }
  }
}

// S4: inverse DFT along x. One block per (b,n). M=128 [Ur|Ui], K=64, N=c.
// TDM stages 64(k) x 64(c) tile (contiguous region), LDS stride 65.
__global__ __launch_bounds__(256) void k_s4(const float* __restrict__ O,
                                            float* __restrict__ U,
                                            const __bf16* __restrict__ Apk) {
  __shared__ float tile[64 * 65];
  int wave = threadIdx.x >> 5, lane = threadIdx.x & 31;
  int b = blockIdx.x >> 4, n = blockIdx.x & 15;
  const float* Ob = O + (b * 16 + n) * 4096;  // + k*64 + c
#if HAVE_TDM
  if (threadIdx.x < 32) {
    tdm_load_2d(tile, Ob, 64, 64, 64, 5);
    __builtin_amdgcn_s_wait_tensorcnt(0);
  }
#else
#pragma unroll
  for (int i = 0; i < 16; ++i) {
    int idx = threadIdx.x + i * 256;
    tile[(idx >> 6) * 65 + (idx & 63)] = Ob[idx];
  }
#endif
  __syncthreads();
#pragma unroll
  for (int r = 0; r < 4; ++r) {
    int wt = wave + r * 8;
    int mt = wt & 7, nt = wt >> 3;
    int c = nt * 16 + (lane & 15);
    v8f acc = {};
#pragma unroll
    for (int ks = 0; ks < 2; ++ks) {
      v16bf a = *(const v16bf*)(Apk + ((mt * 2 + ks) * 32 + lane) * 16);
      int kb = ks * 32 + ((lane >> 4) << 4);
      v16bf bv;
#pragma unroll
      for (int e = 0; e < 16; ++e) bv[e] = (__bf16)tile[(kb + e) * 65 + c];
      acc = WMMA_BF16(a, bv, acc);
    }
    int r0 = mt * 16 + ((lane >> 4) << 3);
#pragma unroll
    for (int j = 0; j < 8; ++j) {
      int row = r0 + j;
      int ri = row >> 6, xx = row & 63;
      U[((b * 64 + c) * 64 + xx) * 32 + ri * 16 + n] = acc[j];
    }
  }
}

// S5: inverse DFT along y, accumulate into Y (+ sbias). K=32 [Ur|Ui], M=y.
// TDM stages 32(cols) x 32(k) tile (contiguous region), LDS stride 33.
__global__ __launch_bounds__(256) void k_s5(const float* __restrict__ U,
                                            float* __restrict__ Y,
                                            const __bf16* __restrict__ Apk,
                                            const float* __restrict__ sbias) {
  __shared__ float tile[32 * 33];
  int wave = threadIdx.x >> 5, lane = threadIdx.x & 31;
  int col0 = blockIdx.x * 32;                  // (b*64+c)*64 + x
  const float* src = U + (size_t)col0 * 32;
#if HAVE_TDM
  if (threadIdx.x < 32) {
    tdm_load_2d(tile, src, 32, 32, 32, 4);
    __builtin_amdgcn_s_wait_tensorcnt(0);
  }
#else
#pragma unroll
  for (int i = 0; i < 4; ++i) {
    int idx = threadIdx.x + i * 256;
    tile[(idx >> 5) * 33 + (idx & 31)] = src[idx];
  }
#endif
  __syncthreads();
  int mt = wave & 3, nt = wave >> 2;
  int cl = nt * 16 + (lane & 15);
  int col = col0 + cl;
  v16bf a = *(const v16bf*)(Apk + (mt * 32 + lane) * 16);
  int kb = ((lane >> 4) << 4);
  v16bf bv;
#pragma unroll
  for (int e = 0; e < 16; ++e) bv[e] = (__bf16)tile[cl * 33 + kb + e];
  v8f acc = {};
  acc = WMMA_BF16(a, bv, acc);
  float sb = sbias[(col >> 6) & 63];
  int r0 = mt * 16 + ((lane >> 4) << 3);
#pragma unroll
  for (int j = 0; j < 8; ++j) {
    int y = r0 + j;
    Y[col * 64 + y] += acc[j] + sb;
  }
}

// ---------------------------------------------------------------------------
// Elementwise kernels
// ---------------------------------------------------------------------------

__global__ __launch_bounds__(256) void k_lift(const float* __restrict__ x,
                                              const float* __restrict__ lw,
                                              const float* __restrict__ lb,
                                              float* __restrict__ v) {
  int idx = blockIdx.x * 256 + threadIdx.x;   // over B*T*SP
  int p = idx & 4095, bt = idx >> 12;
  int b = bt >> 4, t = bt & 15;
  float xi = x[idx];
  float gx = (float)(p >> 6) * (1.0f / 63.0f);
  float gy = (float)(p & 63) * (1.0f / 63.0f);
  float* vb = v + ((size_t)(t * NB + b) * NC) * SP + p;   // layout (t,b,c,p)
#pragma unroll 4
  for (int c = 0; c < 64; ++c)
    vb[c * SP] = xi * lw[c] + gx * lw[64 + c] + gy * lw[128 + c] + lb[c];
}

__global__ __launch_bounds__(256) void k_fill(float* __restrict__ dst,
                                              const float* __restrict__ s, int n) {
  int i = blockIdx.x * 256 + threadIdx.x;
  if (i < n) dst[i] = s[0];
}

__global__ __launch_bounds__(256) void k_gate(const float* __restrict__ A,
                                              const float* __restrict__ Bv,
                                              const float* __restrict__ bsc,
                                              float* __restrict__ Zo) {
  int i = blockIdx.x * 256 + threadIdx.x;
  float t = A[i] + Bv[i] + bsc[0];
  Zo[i] = 1.f / (1.f + expf(-t));
}

__global__ __launch_bounds__(256) void k_rh(const float* __restrict__ R,
                                            const float* __restrict__ H,
                                            float* __restrict__ RH) {
  int i = blockIdx.x * 256 + threadIdx.x;
  RH[i] = R[i] * H[i];
}

__global__ __launch_bounds__(256) void k_hup(const float* __restrict__ A,
                                             const float* __restrict__ Bv,
                                             const float* __restrict__ b3,
                                             const float* __restrict__ Zg,
                                             const float* __restrict__ Hin,
                                             float* __restrict__ Hout) {
  int i = blockIdx.x * 256 + threadIdx.x;
  float t = A[i] + Bv[i] + b3[0];
  const float kS = 1.0507009873554805f, kA = 1.6732632423543772f;
  float hh = (t > 0.f) ? kS * t : kS * kA * (expf(t) - 1.f);
  float z = Zg[i];
  Hout[i] = (1.f - z) * Hin[i] + z * hh;
}

__global__ __launch_bounds__(256) void k_addres(float* __restrict__ v,
                                                const float* __restrict__ seq) {
  int i = blockIdx.x * 256 + threadIdx.x;
  v[i] += seq[i];
}

__global__ __launch_bounds__(256) void k_proj(const float* __restrict__ H,
                                              const float* __restrict__ pw,
                                              const float* __restrict__ pb,
                                              float* __restrict__ out) {
  int i = blockIdx.x * 256 + threadIdx.x;   // b*4096 + p
  if (i >= NB * SP) return;
  int b = i >> 12, p = i & 4095;
  const float* hb = H + (size_t)b * NC * SP + p;
  float acc = pb[0];
#pragma unroll 4
  for (int c = 0; c < 64; ++c) acc += hb[c * SP] * pw[c];
  out[i] = acc;
}

// ---------------------------------------------------------------------------
// Host orchestration
// ---------------------------------------------------------------------------

struct FSet { const float *sbias, *cb; const __bf16 *ApkPW, *ApkS3; };

static void fourier_apply(const float* X, float* Y, const FSet& f,
                          float* Tb, float* Zb, float* Ob, float* Ub,
                          const __bf16* A1, const __bf16* A2,
                          const __bf16* A4, const __bf16* A5, hipStream_t s) {
  k_pw<<<dim3(1024), dim3(256), 0, s>>>(X, Y, f.ApkPW, f.cb);
  k_s1<<<dim3(512),  dim3(256), 0, s>>>(X, Tb, A1);
  k_s2<<<dim3(128),  dim3(256), 0, s>>>(Tb, Zb, A2);
  k_s3<<<dim3(512),  dim3(256), 0, s>>>(Zb, Ob, f.ApkS3);
  k_s4<<<dim3(128),  dim3(256), 0, s>>>(Ob, Ub, A4);
  k_s5<<<dim3(1024), dim3(256), 0, s>>>(Ub, Y, A5, f.sbias);
}

extern "C" void kernel_launch(void* const* d_in, const int* in_sizes, int n_in,
                              void* d_out, int out_size, void* d_ws, size_t ws_size,
                              hipStream_t stream) {
  (void)in_sizes; (void)n_in; (void)out_size; (void)ws_size;
  auto P = [&](int i) { return (const float*)d_in[i]; };
  // dict-insertion-order flattening of setup_inputs():
  // 0:x 1:lift_w 2:lift_b 3:proj_w 4:proj_b, then per layer (46 tensors):
  //   f1..f6: swr1,swi1,swr2,swi2,sbias,cw,cb ; then b1,b2,b3,bias_h
  const float* xin    = P(0);
  const float* lift_w = P(1);
  const float* lift_b = P(2);
  const float* proj_w = P(3);
  const float* proj_b = P(4);

  // Workspace carve-up
  char* wp = (char*)d_ws;
  auto alloc = [&](size_t bytes) -> void* {
    void* p = (void*)wp; wp += (bytes + 255) & ~(size_t)255; return p;
  };
  float* v    = (float*)alloc((size_t)VTOT * 4);    // (t,b,c,p)
  float* seq  = (float*)alloc((size_t)VTOT * 4);
  float* h0   = (float*)alloc((size_t)TS * 4);
  float* h1   = (float*)alloc((size_t)TS * 4);
  float* bufA = (float*)alloc((size_t)TS * 4);
  float* bufB = (float*)alloc((size_t)TS * 4);
  float* bufZ = (float*)alloc((size_t)TS * 4);
  float* bufR = (float*)alloc((size_t)TS * 4);
  float* bufRH= (float*)alloc((size_t)TS * 4);
  float* Tb   = (float*)alloc((size_t)1048576 * 4);
  float* Zb   = (float*)alloc((size_t)524288 * 4);
  float* Ob   = (float*)alloc((size_t)524288 * 4);
  float* Ub   = (float*)alloc((size_t)1048576 * 4);
  __bf16* A1  = (__bf16*)alloc(2048 * 2);
  __bf16* A2  = (__bf16*)alloc(8192 * 2);
  __bf16* A4  = (__bf16*)alloc(8192 * 2);
  __bf16* A5  = (__bf16*)alloc(2048 * 2);
  __bf16* ApkPW = (__bf16*)alloc((size_t)12 * 4096 * 2);
  __bf16* ApkS3 = (__bf16*)alloc((size_t)12 * S3PACK * 2);

  // DFT coefficient packs (parameter independent)
  k_prep_dft<<<dim3(8),  dim3(256), 0, stream>>>(A1, 1, 2, 2048);
  k_prep_dft<<<dim3(32), dim3(256), 0, stream>>>(A2, 2, 4, 8192);
  k_prep_dft<<<dim3(32), dim3(256), 0, stream>>>(A4, 4, 2, 8192);
  k_prep_dft<<<dim3(8),  dim3(256), 0, stream>>>(A5, 5, 1, 2048);

  // Weight packs for all 12 fourier-param sets (reused across 16 timesteps)
  FSet fs[12];
  for (int st = 0; st < 12; ++st) {
    int L = st / 6, fi = st % 6;
    int bi = 5 + L * 46 + fi * 7;
    fs[st].sbias = P(bi + 4);
    fs[st].cb    = P(bi + 6);
    fs[st].ApkPW = ApkPW + (size_t)st * 4096;
    fs[st].ApkS3 = ApkS3 + (size_t)st * S3PACK;
    k_prep_pw<<<dim3(16),    dim3(256), 0, stream>>>(P(bi + 5), ApkPW + (size_t)st * 4096);
    k_prep_s3<<<dim3(24576), dim3(256), 0, stream>>>(P(bi + 0), P(bi + 1), P(bi + 2), P(bi + 3),
                                                     ApkS3 + (size_t)st * S3PACK);
  }

  // Lift: v[t,b,c,p]
  k_lift<<<dim3(2048), dim3(256), 0, stream>>>(xin, lift_w, lift_b, v);

  auto cellstep = [&](const float* xslot, const float* hprev, float* hout,
                      int L, const float* b1, const float* b2, const float* b3) {
    fourier_apply(xslot, bufA, fs[L*6+0], Tb, Zb, Ob, Ub, A1, A2, A4, A5, stream);
    fourier_apply(hprev, bufB, fs[L*6+1], Tb, Zb, Ob, Ub, A1, A2, A4, A5, stream);
    k_gate<<<dim3(8192), dim3(256), 0, stream>>>(bufA, bufB, b1, bufZ);
    fourier_apply(xslot, bufA, fs[L*6+2], Tb, Zb, Ob, Ub, A1, A2, A4, A5, stream);
    fourier_apply(hprev, bufB, fs[L*6+3], Tb, Zb, Ob, Ub, A1, A2, A4, A5, stream);
    k_gate<<<dim3(8192), dim3(256), 0, stream>>>(bufA, bufB, b2, bufR);
    k_rh<<<dim3(8192), dim3(256), 0, stream>>>(bufR, hprev, bufRH);
    fourier_apply(xslot, bufA, fs[L*6+4], Tb, Zb, Ob, Ub, A1, A2, A4, A5, stream);
    fourier_apply(bufRH, bufB, fs[L*6+5], Tb, Zb, Ob, Ub, A1, A2, A4, A5, stream);
    k_hup<<<dim3(8192), dim3(256), 0, stream>>>(bufA, bufB, b3, bufZ, hprev, hout);
  };

  // Layer 0 (return sequences), hidden states written directly into seq slots
  {
    int cb = 5 + 0 * 46;
    const float *b1 = P(cb + 42), *b2 = P(cb + 43), *b3 = P(cb + 44), *bh = P(cb + 45);
    k_fill<<<dim3(8192), dim3(256), 0, stream>>>(h0, bh, TS);
    const float* hp = h0;
    for (int t = 0; t < NT; ++t) {
      float* hout = seq + (size_t)t * TS;
      cellstep(v + (size_t)t * TS, hp, hout, 0, b1, b2, b3);
      hp = hout;
    }
    k_addres<<<dim3(131072), dim3(256), 0, stream>>>(v, seq);   // residual
  }

  // Layer 1 (final hidden only), ping-pong h buffers
  {
    int cb = 5 + 1 * 46;
    const float *b1 = P(cb + 42), *b2 = P(cb + 43), *b3 = P(cb + 44), *bh = P(cb + 45);
    k_fill<<<dim3(8192), dim3(256), 0, stream>>>(h0, bh, TS);
    float* hp = h0; float* hn = h1;
    for (int t = 0; t < NT; ++t) {
      cellstep(v + (size_t)t * TS, hp, hn, 1, b1, b2, b3);
      float* tmp = hp; hp = hn; hn = tmp;
    }
    k_proj<<<dim3(128), dim3(256), 0, stream>>>(hp, proj_w, proj_b, (float*)d_out);
  }
}
